// EdgeAwareMultiHeadGraphAttention_16286515986913
// MI455X (gfx1250) — matrix operand order repaired and checked by reading, not compiled, not run on hardware
//
#include <hip/hip_runtime.h>
#include <hip/hip_bf16.h>

// ---------------------------------------------------------------------------
// CDNA5 (gfx1250) edge-aware multi-head graph attention.
//   N=50000 nodes, E=300000 edges, C=256, H=8, D=32   (sizes derived at launch)
// All dense GEMMs run as bf16 WMMA (v_wmma_f32_16x16x32_bf16).  fp32->bf16
// conversion is hoisted into one-shot prep kernels so the GEMM K-loop is
// pure loads + WMMA.  Segment softmax / scatter use float atomics.
// ---------------------------------------------------------------------------

typedef __attribute__((ext_vector_type(16))) __bf16 v16bf;
typedef __attribute__((ext_vector_type(8)))  float  v8f;

union FragBF {
    v16bf        bf;
    unsigned int u[8];
    uint4        q4[2];
};

// round-half-up fp32 -> bf16 (single)
static __device__ __forceinline__ unsigned short f2bf(float f) {
    return (unsigned short)((__float_as_uint(f) + 0x8000u) >> 16);
}
// round-half-up fp32 pair -> packed bf16x2 via single v_perm_b32
static __device__ __forceinline__ unsigned pack2(float lo, float hi) {
    unsigned ulo = __float_as_uint(lo) + 0x8000u;
    unsigned uhi = __float_as_uint(hi) + 0x8000u;
    return __builtin_amdgcn_perm(uhi, ulo, 0x07060302u);
}

// float atomic max via signed/unsigned integer ordering trick (-inf init)
static __device__ __forceinline__ void atomicMaxFloat(float* addr, float val) {
    if (val >= 0.0f)
        atomicMax((int*)addr, __float_as_int(val));
    else
        atomicMin((unsigned int*)addr, __float_as_uint(val));
}

// ---------------------------------------------------------------------------
// prep: fp32 [n8*8] -> bf16, 8 elements per thread, fully coalesced
// ---------------------------------------------------------------------------
__global__ __launch_bounds__(256) void cvt_bf16_kernel(
    const float* __restrict__ src, unsigned short* __restrict__ dst, int n8)
{
    const int idx = blockIdx.x * 256 + threadIdx.x;
    if (idx >= n8) return;
    const float4* s = reinterpret_cast<const float4*>(src) + 2 * (size_t)idx;
    const float4 f0 = s[0];
    const float4 f1 = s[1];
    uint4 o;
    o.x = pack2(f0.x, f0.y);
    o.y = pack2(f0.z, f0.w);
    o.z = pack2(f1.x, f1.y);
    o.w = pack2(f1.z, f1.w);
    reinterpret_cast<uint4*>(dst)[idx] = o;
}

// ---------------------------------------------------------------------------
// prep: W[256,256] fp32 row-major -> Wt[256,256] bf16 "column-major" (n,k)
// ---------------------------------------------------------------------------
__global__ __launch_bounds__(256) void wt_kernel(
    const float* __restrict__ W, unsigned short* __restrict__ Wt)
{
    const int idx = blockIdx.x * 256 + threadIdx.x;   // idx = n*256 + k
    const int n = idx >> 8;
    const int k = idx & 255;
    Wt[idx] = f2bf(W[k * 256 + n]);
}

// ---------------------------------------------------------------------------
// Out[M,256] = Abf[M,256](bf16) @ Wt(bf16, col-major) + bias(fp32)
// Block: 128 threads (4 waves).  Block tile: 128 rows x 64 cols.
// Wave w: rows m0+32w..+31 (two 16-row A frags), four 16x16 N-tiles.
// K-loop: 4 global b128 (A) + 8 ds b128 (B) + 8 WMMA; no conversion VALU.
// ---------------------------------------------------------------------------
__global__ __launch_bounds__(128) void gemm256_bf16wmma(
    const unsigned short* __restrict__ Abf, const unsigned short* __restrict__ Wt,
    const float* __restrict__ bias, float* __restrict__ Out, int M)
{
    __shared__ unsigned short Wl[64 * 256];   // 32 KB: W panel, column-major bf16

    const int tid  = threadIdx.x;
    const int lane = tid & 31;
    const int wv   = tid >> 5;
    const int m0   = blockIdx.x * 128;
    const int n0   = blockIdx.y * 64;

    // Stage 64-row panel of Wt into LDS: straight uint4 memcpy (coalesced).
    {
        const uint4* s = reinterpret_cast<const uint4*>(Wt + (size_t)n0 * 256);
        uint4*       d = reinterpret_cast<uint4*>(Wl);
        for (int i = tid; i < 2048; i += 128) d[i] = s[i];
    }
    __syncthreads();

    v8f zero = {};
    v8f acc[2][4];
    #pragma unroll
    for (int m = 0; m < 2; ++m)
        #pragma unroll
        for (int t = 0; t < 4; ++t) acc[m][t] = zero;

    int row0 = m0 + 32 * wv + (lane & 15);
    int row1 = row0 + 16;
    if (row0 > M - 1) row0 = M - 1;                 // clamp for tail blocks
    if (row1 > M - 1) row1 = M - 1;
    const unsigned short* ar0 = Abf + (size_t)row0 * 256;
    const unsigned short* ar1 = Abf + (size_t)row1 * 256;

    // A layout (matches B): lanes 0-15 carry K 0..15, lanes 16-31 K 16..31
    const int offA = (lane & 16) ? 16 : 0;
    const int kbB  = (lane & 16) ? 16 : 0;          // B: lanes 16-31 carry K 16..31
    const int colL = lane & 15;

    #pragma unroll
    for (int kc = 0; kc < 256; kc += 32) {
        __builtin_prefetch(ar0 + kc + 64, 0, 1);    // global_prefetch_b8 (speculative)

        FragBF a0, a1;
        a0.q4[0] = *reinterpret_cast<const uint4*>(ar0 + kc + offA);
        a0.q4[1] = *reinterpret_cast<const uint4*>(ar0 + kc + offA + 8);
        a1.q4[0] = *reinterpret_cast<const uint4*>(ar1 + kc + offA);
        a1.q4[1] = *reinterpret_cast<const uint4*>(ar1 + kc + offA + 8);

        #pragma unroll
        for (int t = 0; t < 4; ++t) {
            FragBF b;
            const uint4* p = reinterpret_cast<const uint4*>(
                &Wl[(t * 16 + colL) * 256 + kc + kbB]);
            b.q4[0] = p[0];
            b.q4[1] = p[1];
            acc[0][t] = __builtin_amdgcn_wmma_f32_16x16x32_bf16(
                false, a0.bf, false, b.bf, (short)0, acc[0][t], false, false);
            acc[1][t] = __builtin_amdgcn_wmma_f32_16x16x32_bf16(
                false, a1.bf, false, b.bf, (short)0, acc[1][t], false, false);
        }
    }

    // C/D layout: VGPR v -> (row = v + 8*(lane>=16), col = lane%16)
    const int rbase = m0 + 32 * wv + ((lane & 16) >> 1);
    #pragma unroll
    for (int t = 0; t < 4; ++t) {
        const int col = n0 + t * 16 + colL;
        const float bv = bias[col];
        #pragma unroll
        for (int m = 0; m < 2; ++m) {
            #pragma unroll
            for (int v = 0; v < 8; ++v) {
                const int r = rbase + 16 * m + v;
                if (r < M) Out[(size_t)r * 256 + col] = acc[m][t][v] + bv;
            }
        }
    }
}

// ---------------------------------------------------------------------------
// init: aggregated=0, smax=-inf, denom=0
// ---------------------------------------------------------------------------
__global__ __launch_bounds__(256) void init_kernel(
    float* __restrict__ agg, float* __restrict__ smax, float* __restrict__ denom, int N)
{
    const int idx = blockIdx.x * 256 + threadIdx.x;
    if (idx < N * 256) agg[idx] = 0.0f;
    if (idx < N * 8) {
        smax[idx]  = -3.402823466e38f;
        denom[idx] = 0.0f;
    }
}

// ---------------------------------------------------------------------------
// scores: wave per edge; lane = d; per head: reduce q[src]*k[dst]+esh*edge
// then atomic segment-max into smax[src][h].
// ---------------------------------------------------------------------------
__global__ __launch_bounds__(256) void scores_kernel(
    const float* __restrict__ q, const float* __restrict__ k,
    const float* __restrict__ esh, const float* __restrict__ edge_states,
    const long long* __restrict__ ei, float* __restrict__ scores,
    float* __restrict__ smax, int E)
{
    const int lane = threadIdx.x & 31;
    const int e    = blockIdx.x * 8 + (threadIdx.x >> 5);
    if (e >= E) return;
    const int src = (int)ei[e];
    const int dst = (int)ei[(size_t)E + e];
    const float* qp = q + (size_t)src * 256;
    const float* kp = k + (size_t)dst * 256;
    const float* ep = esh + (size_t)e * 256;
    const float* xp = edge_states + (size_t)e * 256;

    #pragma unroll
    for (int h = 0; h < 8; ++h) {
        const int c = h * 32 + lane;
        float p = qp[c] * kp[c] + ep[c] * xp[c];
        #pragma unroll
        for (int o = 16; o > 0; o >>= 1) p += __shfl_xor(p, o, 32);
        p *= 0.17677669529663687f;              // 1/sqrt(32)
        if (lane == 0) {
            scores[(size_t)e * 8 + h] = p;
            atomicMaxFloat(&smax[src * 8 + h], p);
        }
    }
}

// ---------------------------------------------------------------------------
// expv: scores -> exp(scores - smax[src]);  denom[src][h] += expv
// ---------------------------------------------------------------------------
__global__ __launch_bounds__(256) void expv_kernel(
    float* __restrict__ sc, const float* __restrict__ smax,
    float* __restrict__ denom, const long long* __restrict__ ei, int E)
{
    const int idx = blockIdx.x * 256 + threadIdx.x;
    if (idx >= E * 8) return;
    const int e = idx >> 3;
    const int h = idx & 7;
    const int src = (int)ei[e];
    const float ev = __expf(sc[idx] - smax[src * 8 + h]);
    sc[idx] = ev;
    atomicAdd(&denom[src * 8 + h], ev);
}

// ---------------------------------------------------------------------------
// aggregate: wave per edge.  att[h]=expv/denom -> out_att; scatter
// att*v[dst]*node_states[dst] into aggregated[src] with float atomics.
// ---------------------------------------------------------------------------
__global__ __launch_bounds__(256) void aggregate_kernel(
    const float* __restrict__ expv, const float* __restrict__ denom,
    const float* __restrict__ v, const float* __restrict__ node_states,
    const long long* __restrict__ ei, float* __restrict__ agg,
    float* __restrict__ out_att, int E)
{
    const int lane = threadIdx.x & 31;
    const int e    = blockIdx.x * 8 + (threadIdx.x >> 5);
    if (e >= E) return;
    const int src = (int)ei[e];
    const int dst = (int)ei[(size_t)E + e];

    float a_att = 0.0f;
    if (lane < 8) {
        a_att = expv[(size_t)e * 8 + lane] / (denom[src * 8 + lane] + 1e-12f);
        out_att[(size_t)e * 8 + lane] = a_att;
    }

    const float* vp = v + (size_t)dst * 256;
    const float* np = node_states + (size_t)dst * 256;
    float* ap = agg + (size_t)src * 256;

    #pragma unroll
    for (int h = 0; h < 8; ++h) {
        const float ah = __shfl(a_att, h, 32);
        const int c = h * 32 + lane;
        atomicAdd(&ap[c], ah * vp[c] * np[c]);
    }
}

// ---------------------------------------------------------------------------
// edge_update: [E,8] @ Weo[8,256] + beo   (K=8; bandwidth-bound on the store)
// ---------------------------------------------------------------------------
__global__ __launch_bounds__(256) void edge_out_kernel(
    const float* __restrict__ att, const float* __restrict__ Weo,
    const float* __restrict__ beo, float* __restrict__ out_edge, int E)
{
    __shared__ float att_s[8];
    const int e = blockIdx.x;
    const int c = threadIdx.x;
    if (c < 8) att_s[c] = att[(size_t)e * 8 + c];
    __syncthreads();
    float s = beo[c];
    #pragma unroll
    for (int h = 0; h < 8; ++h) s += att_s[h] * Weo[h * 256 + c];
    out_edge[(size_t)e * 256 + c] = s;
}

// ---------------------------------------------------------------------------
extern "C" void kernel_launch(void* const* d_in, const int* in_sizes, int n_in,
                              void* d_out, int out_size, void* d_ws, size_t ws_size,
                              hipStream_t stream)
{
    const float*     node_states = (const float*)d_in[0];
    const long long* edge_index  = (const long long*)d_in[1];
    const float*     edge_states = (const float*)d_in[2];
    const float* Wq  = (const float*)d_in[3];  const float* bq  = (const float*)d_in[4];
    const float* Wk  = (const float*)d_in[5];  const float* bk  = (const float*)d_in[6];
    const float* Wv  = (const float*)d_in[7];  const float* bv  = (const float*)d_in[8];
    const float* We  = (const float*)d_in[9];  const float* be  = (const float*)d_in[10];
    const float* Wno = (const float*)d_in[11]; const float* bno = (const float*)d_in[12];
    const float* Weo = (const float*)d_in[13]; const float* beo = (const float*)d_in[14];

    const int N = in_sizes[0] / 256;
    const int E = in_sizes[2] / 256;

    // workspace carve-up
    float* ws    = (float*)d_ws;
    float* q     = ws;
    float* k     = q     + (size_t)N * 256;
    float* v     = k     + (size_t)N * 256;
    float* esh   = v     + (size_t)N * 256;
    float* sc    = esh   + (size_t)E * 256;   // scores, then expv in-place
    float* smax  = sc    + (size_t)E * 8;
    float* denom = smax  + (size_t)N * 8;
    float* agg   = denom + (size_t)N * 8;
    unsigned short* nodes_bf = (unsigned short*)(agg + (size_t)N * 256);
    unsigned short* edges_bf = nodes_bf + (size_t)N * 256;
    unsigned short* agg_bf   = edges_bf + (size_t)E * 256;
    unsigned short* Wt_q     = agg_bf + (size_t)N * 256;
    unsigned short* Wt_k     = Wt_q  + 65536;
    unsigned short* Wt_v     = Wt_k  + 65536;
    unsigned short* Wt_e     = Wt_v  + 65536;
    unsigned short* Wt_no    = Wt_e  + 65536;

    float* out_node = (float*)d_out;
    float* out_edge = out_node + (size_t)N * 256;
    float* out_att  = out_edge + (size_t)E * 256;

    const dim3 gN((N + 127) / 128, 4);
    const dim3 gE((E + 127) / 128, 4);
    const int  gW = 65536 / 256;              // weight transpose grid

    init_kernel<<<(N * 256 + 255) / 256, 256, 0, stream>>>(agg, smax, denom, N);

    // one-shot conversions (hoisted out of the GEMMs)
    cvt_bf16_kernel<<<(N * 32 + 255) / 256, 256, 0, stream>>>(node_states, nodes_bf, N * 32);
    cvt_bf16_kernel<<<(E * 32 + 255) / 256, 256, 0, stream>>>(edge_states, edges_bf, E * 32);
    wt_kernel<<<gW, 256, 0, stream>>>(Wq,  Wt_q);
    wt_kernel<<<gW, 256, 0, stream>>>(Wk,  Wt_k);
    wt_kernel<<<gW, 256, 0, stream>>>(Wv,  Wt_v);
    wt_kernel<<<gW, 256, 0, stream>>>(We,  Wt_e);
    wt_kernel<<<gW, 256, 0, stream>>>(Wno, Wt_no);

    gemm256_bf16wmma<<<gN, 128, 0, stream>>>(nodes_bf, Wt_q, bq, q, N);
    gemm256_bf16wmma<<<gN, 128, 0, stream>>>(nodes_bf, Wt_k, bk, k, N);
    gemm256_bf16wmma<<<gN, 128, 0, stream>>>(nodes_bf, Wt_v, bv, v, N);
    gemm256_bf16wmma<<<gE, 128, 0, stream>>>(edges_bf, Wt_e, be, esh, E);

    scores_kernel<<<(E + 7) / 8, 256, 0, stream>>>(q, k, esh, edge_states,
                                                   edge_index, sc, smax, E);
    expv_kernel<<<(E * 8 + 255) / 256, 256, 0, stream>>>(sc, smax, denom,
                                                         edge_index, E);
    aggregate_kernel<<<(E + 7) / 8, 256, 0, stream>>>(sc, denom, v, node_states,
                                                      edge_index, agg, out_att, E);

    cvt_bf16_kernel<<<(N * 32 + 255) / 256, 256, 0, stream>>>(agg, agg_bf, N * 32);
    gemm256_bf16wmma<<<gN, 128, 0, stream>>>(agg_bf, Wt_no, bno, out_node, N);
    edge_out_kernel<<<E, 256, 0, stream>>>(out_att, Weo, beo, out_edge, E);
}
